// CondConv3d_35699768164362
// MI455X (gfx1250) — matrix operand order repaired and checked by reading, not compiled, not run on hardware
//
#include <hip/hip_runtime.h>
#include <math.h>

typedef __bf16 bf16;
typedef __attribute__((ext_vector_type(4)))  bf16  v4bf;
typedef __attribute__((ext_vector_type(8)))  bf16  v8bf;
typedef __attribute__((ext_vector_type(16))) bf16  v16bf;
typedef __attribute__((ext_vector_type(8)))  float v8f;

#define NB 8
#define CI 32
#define CO 32
#define DD 24
#define HH 64
#define WW 64
#define SPAT (DD*HH*WW)          // 98304
#define TAPS 27
#define EWN (CO*TAPS*CI)         // 27648 effective-weight elems per sample
#define XP 36                    // padded channel stride (halfs) in LDS x tile
#define AP 872                   // padded per-co stride (halfs) in LDS A panel
#define XSH_HALFS (3*6*66*XP)    // 42768 halfs = 85536 B
#define APAN_HALFS (CO*AP)       // 27904 halfs = 55808 B
#define SH_BYTES ((XSH_HALFS + APAN_HALFS) * 2)   // 141344 B dynamic LDS

#define XBF_OFF   524288ull                        // scratch offset of bf16 x copy
#define XBF_BYTES ((size_t)NB*SPAT*CI*2)           // 50,331,648 B
#define WS_NEED   (XBF_OFF + XBF_BYTES)

// ---------------------------------------------------------------------------
// Kernel 1: 45 box-sums per (n,c) (interval sums for mean-of-conv identity).
// ---------------------------------------------------------------------------
__global__ __launch_bounds__(256) void k_boxsum(const float* __restrict__ x,
                                                float* __restrict__ box)
{
    const int nc = blockIdx.x;                      // n*32 + c
    const float* xs = x + (size_t)nc * SPAT;
    float acc[45];
#pragma unroll
    for (int i = 0; i < 45; ++i) acc[i] = 0.f;

    for (int row = threadIdx.x; row < DD*HH; row += 256) {
        const int d = row >> 6, h = row & 63;
        const float* rp = xs + (size_t)row * WW;
        const float4* r4 = (const float4*)rp;
        float s = 0.f;
#pragma unroll
        for (int i = 0; i < 16; ++i) { float4 v = r4[i]; s += v.x + v.y + v.z + v.w; }
        const float first = rp[0], last = rp[63];
        const float sw[3]  = { s - last, s, s - first };
        const bool  hOK[3] = { h <= HH-2, true, h >= 1 };
        const bool  dOK[5] = { d <= DD-3, d <= DD-2, true, d >= 1, d >= 2 };
#pragma unroll
        for (int dv = 0; dv < 5; ++dv) if (dOK[dv])
#pragma unroll
            for (int hv = 0; hv < 3; ++hv) if (hOK[hv])
#pragma unroll
                for (int wv = 0; wv < 3; ++wv)
                    acc[dv*9 + hv*3 + wv] += sw[wv];
    }
#pragma unroll
    for (int i = 0; i < 45; ++i)
        for (int off = 16; off > 0; off >>= 1)
            acc[i] += __shfl_down(acc[i], off, 32);

    __shared__ float sm[8][45];
    const int wv = threadIdx.x >> 5, ln = threadIdx.x & 31;
    if (ln == 0) {
#pragma unroll
        for (int i = 0; i < 45; ++i) sm[wv][i] = acc[i];
    }
    __syncthreads();
    if (threadIdx.x < 45) {
        float s = 0.f;
        for (int w = 0; w < 8; ++w) s += sm[w][threadIdx.x];
        box[(size_t)nc * 45 + threadIdx.x] = s;
    }
}

// ---------------------------------------------------------------------------
// Kernel 2: attention logits + softmax -> att[n][k]
// ---------------------------------------------------------------------------
__global__ __launch_bounds__(64) void k_att(const float* __restrict__ box,
    const float* __restrict__ aw1, const float* __restrict__ ab1,
    const float* __restrict__ aw2, const float* __restrict__ ab2,
    const float* __restrict__ aw3, const float* __restrict__ ab3,
    float* __restrict__ att)
{
    __shared__ float lg[NB][4];
    const int t = threadIdx.x;
    if (t < NB*4) {
        const int n = t >> 2, k = t & 3;
        float acc = 0.f;
        for (int c = 0; c < CI; ++c) {
            const float* B = box + ((size_t)n*CI + c) * 45;
            acc += aw1[k*CI + c] * B[22];
            const float* w2 = aw2 + ((size_t)k*CI + c) * 27;
#pragma unroll
            for (int kd = 0; kd < 3; ++kd)
                for (int kh = 0; kh < 3; ++kh)
                    for (int kw = 0; kw < 3; ++kw)
                        acc += w2[kd*9+kh*3+kw] * B[(kd+1)*9 + kh*3 + kw];
            const float* w3 = aw3 + ((size_t)k*CI + c) * 45;
#pragma unroll
            for (int kd = 0; kd < 5; ++kd)
                for (int kh = 0; kh < 3; ++kh)
                    for (int kw = 0; kw < 3; ++kw)
                        acc += w3[kd*9+kh*3+kw] * B[kd*9 + kh*3 + kw];
        }
        lg[n][k] = ab1[k] + ab2[k] + ab3[k] + acc * (1.f / (float)SPAT);
    }
    __syncthreads();
    if (t < NB) {
        float m = lg[t][0];
        for (int k = 1; k < 4; ++k) m = fmaxf(m, lg[t][k]);
        float e[4], s = 0.f;
        for (int k = 0; k < 4; ++k) { e[k] = expf(lg[t][k] - m); s += e[k]; }
        for (int k = 0; k < 4; ++k) att[t*4 + k] = e[k] / s;
    }
}

// ---------------------------------------------------------------------------
// Kernel 3: effective weight  W_n = conv_w + sum_k att[n,k]*weight[k]
// ---------------------------------------------------------------------------
__global__ __launch_bounds__(256) void k_wmix(const float* __restrict__ att,
    const float* __restrict__ conv_w, const float* __restrict__ weight,
    bf16* __restrict__ ew)
{
    const int n = blockIdx.x;
    const float a0 = att[n*4+0], a1 = att[n*4+1], a2 = att[n*4+2], a3 = att[n*4+3];
    bf16* dst = ew + (size_t)n * EWN;
    for (int e = threadIdx.x; e < EWN; e += 256) {
        const int co  = e / (TAPS*CI);
        const int rem = e - co * (TAPS*CI);
        const int tap = rem >> 5, c = rem & 31;
        const int src = (co*CI + c)*TAPS + tap;
        float v = conv_w[src] + a0*weight[src] + a1*weight[src + EWN]
                              + a2*weight[src + 2*EWN] + a3*weight[src + 3*EWN];
        dst[e] = (bf16)v;
    }
}

// ---------------------------------------------------------------------------
// Kernel 3b: one-time fp32 -> bf16 repack of x into [n][d][h][w][c] (c inner).
// ---------------------------------------------------------------------------
__global__ __launch_bounds__(256) void k_xcvt(const float* __restrict__ x,
                                              bf16* __restrict__ xbf)
{
    const int n  = blockIdx.x / (SPAT/256);
    const int sp = (blockIdx.x % (SPAT/256)) * 256 + threadIdx.x;
    const float* src = x + (size_t)n * CI * SPAT + sp;
    bf16* dst = xbf + ((size_t)n * SPAT + sp) * CI;
#pragma unroll
    for (int j = 0; j < 4; ++j) {
        v8bf v;
#pragma unroll
        for (int k = 0; k < 8; ++k)
            v[k] = (bf16)src[(size_t)(8*j + k) * SPAT];
        *(v8bf*)&dst[8*j] = v;
    }
}

// ---------------------------------------------------------------------------
// Shared GEMM core: 27-tap implicit GEMM over staged LDS tiles.
// Each wave: two 32co x 16w output tiles (A fragments reused), 108 WMMAs.
// ---------------------------------------------------------------------------
__device__ __forceinline__ v16bf ld16b(const bf16* p0)   // 16 contig halfs, 8B-aligned
{
    v4bf a0 = *(const v4bf*)p0, a1 = *(const v4bf*)(p0 + 4);
    v4bf b0 = *(const v4bf*)(p0 + 8), b1 = *(const v4bf*)(p0 + 12);
    v8bf lo = __builtin_shufflevector(a0, a1, 0,1,2,3,4,5,6,7);
    v8bf hi = __builtin_shufflevector(b0, b1, 0,1,2,3,4,5,6,7);
    return __builtin_shufflevector(lo, hi, 0,1,2,3,4,5,6,7,8,9,10,11,12,13,14,15);
}
__device__ __forceinline__ v16bf ld16a(const bf16* p0, const bf16* p1) // 2x 16B-aligned
{
    v8bf lo = *(const v8bf*)p0;
    v8bf hi = *(const v8bf*)p1;
    return __builtin_shufflevector(lo, hi, 0,1,2,3,4,5,6,7,8,9,10,11,12,13,14,15);
}

__device__ __forceinline__ void conv_core(const bf16* xsh, const bf16* apan,
    const float* __restrict__ bias, float* __restrict__ out,
    int n, int d0, int h0, int tid)
{
    const int wv    = tid >> 5, ln = tid & 31;
    const int hrelA = wv >> 2;              // h rows: hrelA and hrelA+2
    const int w0    = (wv & 3) * 16;
    const int lh    = ln & 15;
    const int hi    = ln >> 4;
    const int cA    = hi ? 16 : 0;
    const int cB    = hi ? 16 : 0;

    v8f acc0a = {}, acc1a = {}, acc0b = {}, acc1b = {};

#pragma unroll
    for (int t = 0; t < TAPS; ++t) {
        const int kd = t / 9, r9 = t - kd*9;
        const int kh = r9 / 3, kw = r9 - (r9/3)*3;

        const bf16* a0p = apan + lh * AP + t*CI + cA;
        const bf16* a1p = apan + (16 + lh) * AP + t*CI + cA;
        const v16bf A0 = ld16a(a0p, a0p + 8);
        const v16bf A1 = ld16a(a1p, a1p + 8);

        const int wofs = w0 + lh + kw;
        const v16bf Ba = ld16b(&xsh[(((kd*6) + (hrelA     + kh))*66 + wofs) * XP + cB]);
        const v16bf Bb = ld16b(&xsh[(((kd*6) + (hrelA + 2 + kh))*66 + wofs) * XP + cB]);

        acc0a = __builtin_amdgcn_wmma_f32_16x16x32_bf16(false, A0, false, Ba,
                                                        (short)0, acc0a, false, false);
        acc1a = __builtin_amdgcn_wmma_f32_16x16x32_bf16(false, A1, false, Ba,
                                                        (short)0, acc1a, false, false);
        acc0b = __builtin_amdgcn_wmma_f32_16x16x32_bf16(false, A0, false, Bb,
                                                        (short)0, acc0b, false, false);
        acc1b = __builtin_amdgcn_wmma_f32_16x16x32_bf16(false, A1, false, Bb,
                                                        (short)0, acc1b, false, false);
    }

    const int wo = w0 + lh;
    const int cobase = hi ? 8 : 0;
    float* opa = out + (size_t)n * CO * SPAT + ((size_t)d0 * HH + (h0 + hrelA)) * WW + wo;
    float* opb = opa + 2 * WW;
#pragma unroll
    for (int r = 0; r < 8; ++r) {
        const int co = cobase + r;
        const float b0 = bias[co], b1 = bias[co + 16];
        opa[(size_t)co * SPAT]        = acc0a[r] + b0;
        opa[(size_t)(co + 16) * SPAT] = acc1a[r] + b1;
        opb[(size_t)co * SPAT]        = acc0b[r] + b0;
        opb[(size_t)(co + 16) * SPAT] = acc1b[r] + b1;
    }
}

// stage full bf16 A panel (55296 B) via async global->LDS DMA, padded stride
__device__ __forceinline__ void stage_apan_async(const bf16* ew, bf16* apan,
                                                 int n, int tid)
{
    const unsigned apan_lds = (unsigned)(uintptr_t)apan;
    const unsigned long long gbase =
        (unsigned long long)(uintptr_t)(ew) + (size_t)n * EWN * 2;
#pragma unroll
    for (int it = 0; it < 27; ++it) {                 // 27*256 x 8B = 55296 B
        const int i  = tid + it * 256;
        const int co = i / 216, j = i - co * 216;
        const unsigned ldsa = apan_lds + (unsigned)(co * (AP*2) + j * 8);
        const unsigned goff = (unsigned)(co * (TAPS*CI*2) + j * 8);
        asm volatile("global_load_async_to_lds_b64 %0, %1, %2"
                     :: "v"(ldsa), "v"(goff), "s"(gbase) : "memory");
    }
}

// ---------------------------------------------------------------------------
// Kernel 4a (fast): x tile staged by async DMA from pre-converted bf16 x.
// ---------------------------------------------------------------------------
__global__ __launch_bounds__(256) void k_conv_fast(const bf16* __restrict__ xbf,
    const bf16* __restrict__ ew, const float* __restrict__ bias,
    float* __restrict__ out)
{
    extern __shared__ __align__(16) char smem[];
    bf16* xsh  = (bf16*)smem;                         // [dd3][hh6][ww66][cXP]
    bf16* apan = xsh + XSH_HALFS;                     // [co32][AP]

    const int n = blockIdx.z, d0 = blockIdx.y, h0 = blockIdx.x * 4;
    const int tid = threadIdx.x;

    stage_apan_async(ew, apan, n, tid);

    // x halo tile: 1188 positions x 64B, async 8B chunks; zero-fill borders
    const unsigned xsh_lds = (unsigned)(uintptr_t)xsh;
    const unsigned long long xbase =
        (unsigned long long)(uintptr_t)(xbf) + (size_t)n * SPAT * CI * 2;
    for (int idx = tid; idx < 3*6*66*8; idx += 256) {
        const int pos = idx >> 3, j = idx & 7;
        const int dd = pos / (6*66);
        const int r  = pos - dd * (6*66);
        const int hh = r / 66, ww = r - hh * 66;
        const int gd = d0 - 1 + dd, gh = h0 - 1 + hh, gw = ww - 1;
        if (gd >= 0 && gd < DD && gh >= 0 && gh < HH && gw >= 0 && gw < WW) {
            const unsigned ldsa = xsh_lds + (unsigned)(pos * (XP*2) + j * 8);
            const unsigned goff = (unsigned)((((gd*HH + gh)*WW + gw) * CI + j*4) * 2);
            asm volatile("global_load_async_to_lds_b64 %0, %1, %2"
                         :: "v"(ldsa), "v"(goff), "s"(xbase) : "memory");
        } else {
            v4bf z = {};
            *(v4bf*)&xsh[pos*XP + j*4] = z;
        }
    }

    asm volatile("s_wait_asynccnt 0" ::: "memory");
    __syncthreads();
    conv_core(xsh, apan, bias, out, n, d0, h0, tid);
}

// ---------------------------------------------------------------------------
// Kernel 4b (fallback, small scratch): convert fp32 x while staging.
// ---------------------------------------------------------------------------
__global__ __launch_bounds__(256) void k_conv(const float* __restrict__ x,
    const bf16* __restrict__ ew, const float* __restrict__ bias,
    float* __restrict__ out)
{
    extern __shared__ __align__(16) char smem[];
    bf16* xsh  = (bf16*)smem;
    bf16* apan = xsh + XSH_HALFS;

    const int n = blockIdx.z, d0 = blockIdx.y, h0 = blockIdx.x * 4;
    const int tid = threadIdx.x;
    const float* xb = x + (size_t)n * CI * SPAT;

    stage_apan_async(ew, apan, n, tid);

    for (int pos = tid; pos < 3*6*66; pos += 256) {
        const int dd = pos / (6*66);
        const int r  = pos - dd * (6*66);
        const int hh = r / 66, ww = r - hh * 66;
        const int gd = d0 - 1 + dd, gh = h0 - 1 + hh, gw = ww - 1;
        if (gd >= 0 && gd < DD && gh >= 0 && gh < HH && gw >= 0 && gw < WW) {
            const float* src = xb + ((size_t)gd * HH + gh) * WW + gw;
#pragma unroll
            for (int j = 0; j < 8; ++j) {
                v4bf v;
                v[0] = (bf16)src[(size_t)(4*j+0) * SPAT];
                v[1] = (bf16)src[(size_t)(4*j+1) * SPAT];
                v[2] = (bf16)src[(size_t)(4*j+2) * SPAT];
                v[3] = (bf16)src[(size_t)(4*j+3) * SPAT];
                *(v4bf*)&xsh[pos*XP + 4*j] = v;
            }
        } else {
            v4bf z = {};
#pragma unroll
            for (int j = 0; j < 8; ++j) *(v4bf*)&xsh[pos*XP + 4*j] = z;
        }
    }

    asm volatile("s_wait_asynccnt 0" ::: "memory");
    __syncthreads();
    conv_core(xsh, apan, bias, out, n, d0, h0, tid);
}

// ---------------------------------------------------------------------------
extern "C" void kernel_launch(void* const* d_in, const int* in_sizes, int n_in,
                              void* d_out, int out_size, void* d_ws, size_t ws_size,
                              hipStream_t stream)
{
    const float* x      = (const float*)d_in[0];
    const float* weight = (const float*)d_in[1];
    const float* conv_w = (const float*)d_in[2];
    const float* conv_b = (const float*)d_in[3];
    const float* aw1    = (const float*)d_in[4];
    const float* ab1    = (const float*)d_in[5];
    const float* aw2    = (const float*)d_in[6];
    const float* ab2    = (const float*)d_in[7];
    const float* aw3    = (const float*)d_in[8];
    const float* ab3    = (const float*)d_in[9];

    float* box = (float*)d_ws;                        // 8*32*45 f32 = 46080 B
    float* att = box + NB*CI*45;                      // 32 f32
    bf16*  ewp = (bf16*)((char*)d_ws + 49152);        // 442368 B
    bf16*  xbf = (bf16*)((char*)d_ws + XBF_OFF);      // 50.3 MB (if available)

    k_boxsum<<<dim3(NB*CI), dim3(256), 0, stream>>>(x, box);
    k_att   <<<dim3(1),     dim3(64),  0, stream>>>(box, aw1, ab1, aw2, ab2, aw3, ab3, att);
    k_wmix  <<<dim3(NB),    dim3(256), 0, stream>>>(att, conv_w, weight, ewp);

    if (ws_size >= WS_NEED) {
        k_xcvt<<<dim3(NB*(SPAT/256)), dim3(256), 0, stream>>>(x, xbf);
        k_conv_fast<<<dim3(HH/4, DD, NB), dim3(256), SH_BYTES, stream>>>(xbf, ewp,
                                                        conv_b, (float*)d_out);
    } else {
        k_conv<<<dim3(HH/4, DD, NB), dim3(256), SH_BYTES, stream>>>(x, ewp,
                                                        conv_b, (float*)d_out);
    }
}
